// GCNNet_37898791420554
// MI455X (gfx1250) — matrix operand order repaired and checked by reading, not compiled, not run on hardware
//
#include <hip/hip_runtime.h>
#include <hip/hip_bf16.h>

typedef __attribute__((ext_vector_type(16))) _Float16 v16h;
typedef __attribute__((ext_vector_type(8)))  float    v8f;

#define BM 128
#define BN 32
#define BK 64
#define GEMM_THREADS 256

// ---------------------------------------------------------------------------
// WMMA GEMM: C[M,Nout] = opt_relu(A[M,K] @ B[K,Nout] + opt_bias)
// A,B,C fp32 row-major in global; converted to f16 in LDS, fp32 accumulate.
// Block = 8 waves; tile 128(M) x 32(N) x 64(K); each wave issues 4 WMMAs per
// staging round (2 k-substeps x 2 N-subtiles, A fragment reused).
// Staging is branch-free: row index clamped (padded C rows never stored),
// K index clamped + cndmask zero-fill (K padding must be exact zero).
// ---------------------------------------------------------------------------
__global__ __launch_bounds__(GEMM_THREADS)
void k_gemm_wmma(const float* __restrict__ A, const float* __restrict__ B,
                 const float* __restrict__ bias, float* __restrict__ C,
                 int M, int K, int Nout, int relu)
{
    __shared__ _Float16 sA[BM][BK + 2];   // 128 x 66 halves
    __shared__ _Float16 sB[BN][BK + 2];   // transposed: sB[n][k]

    const int tid  = threadIdx.x;
    const int lane = tid & 31;            // wave32
    const int wv   = tid >> 5;            // 0..7
    const int m0   = blockIdx.x * BM;
    const int n0   = blockIdx.y * BN;
    const int r    = lane & 15;
    const int grp  = lane >> 4;

    v8f acc0 = {};
    v8f acc1 = {};

    for (int k0 = 0; k0 < K; k0 += BK) {
        // ---- stage A tile (BM x BK): kk fixed per thread, rows vary -> each
        // global read is a coalesced 32-lane burst along K. No branches.
        {
            const int kk  = tid & (BK - 1);
            const int gk  = k0 + kk;
            const int inb = (gk < K);
            const int gkc = inb ? gk : (K - 1);
            #pragma unroll
            for (int j = 0; j < BM * BK / GEMM_THREADS; ++j) {
                int mm = (tid >> 6) + j * (GEMM_THREADS / BK);
                int gm = m0 + mm;
                gm = (gm < M) ? gm : (M - 1);                 // clamp: safe
                float v = A[(size_t)gm * K + gkc];            // unconditional
                sA[mm][kk] = (_Float16)(inb ? v : 0.0f);      // cndmask
            }
        }
        // ---- stage B tile transposed (sB[n][k]): nn fixed per thread,
        // consecutive tids -> consecutive nn -> coalesced along N.
        {
            const int nn = tid & (BN - 1);
            #pragma unroll
            for (int j = 0; j < BK * BN / GEMM_THREADS; ++j) {
                int kk  = (tid >> 5) + j * (GEMM_THREADS / BN);
                int gk  = k0 + kk;
                int inb = (gk < K);
                int gkc = inb ? gk : (K - 1);
                float v = B[(size_t)gkc * Nout + (n0 + nn)];  // unconditional
                sB[nn][kk] = (_Float16)(inb ? v : 0.0f);
            }
        }
        __syncthreads();

        // ---- 2 k-substeps x 2 N-subtiles = 4 WMMAs per barrier pair
        #pragma unroll
        for (int ks = 0; ks < 2; ++ks) {
            const int kb = ks * 32;
            v16h a, b0v, b1v;
            // A frag (16-bit A 16x32): lane holds M=lane&15;
            // halves 0..7 -> K=grp*8+h, halves 8..15 -> K=16+grp*8+(h-8)
            #pragma unroll
            for (int h = 0; h < 8; ++h) {
                a[h]     = sA[wv * 16 + r][kb + grp * 8 + h];
                a[h + 8] = sA[wv * 16 + r][kb + 16 + grp * 8 + h];
            }
            // B frag (16-bit B 32x16): lane holds column N=lane&15;
            // halves 0..7 -> K=grp*8+h, halves 8..15 -> K=16+grp*8+(h-8)
            #pragma unroll
            for (int h = 0; h < 8; ++h) {
                b0v[h]     = sB[r][kb + grp * 8 + h];
                b0v[h + 8] = sB[r][kb + 16 + grp * 8 + h];
                b1v[h]     = sB[16 + r][kb + grp * 8 + h];
                b1v[h + 8] = sB[16 + r][kb + 16 + grp * 8 + h];
            }
            acc0 = __builtin_amdgcn_wmma_f32_16x16x32_f16(false, a, false, b0v,
                                                          (short)0, acc0, false, false);
            acc1 = __builtin_amdgcn_wmma_f32_16x16x32_f16(false, a, false, b1v,
                                                          (short)0, acc1, false, false);
        }
        __syncthreads();
    }

    // Epilogue: C/D layout -> VGPR i: M = grp*8 + i, N = lane&15
    #pragma unroll
    for (int i = 0; i < 8; ++i) {
        int m = m0 + wv * 16 + grp * 8 + i;
        if (m < M) {
            int n   = n0 + r;
            float v = acc0[i];
            if (bias) v += bias[n];
            if (relu) v = fmaxf(v, 0.0f);
            C[(size_t)m * Nout + n] = v;

            n = n0 + 16 + r;
            v = acc1[i];
            if (bias) v += bias[n];
            if (relu) v = fmaxf(v, 0.0f);
            C[(size_t)m * Nout + n] = v;
        }
    }
}

// ---------------------------------------------------------------------------
// Degree / normalization
// ---------------------------------------------------------------------------
__global__ void k_deg_init(int* __restrict__ deg, int n) {
    int i = blockIdx.x * blockDim.x + threadIdx.x;
    if (i < n) deg[i] = 1;                       // self-loop
}
__global__ void k_deg_accum(const int* __restrict__ dst, int* __restrict__ deg, int E) {
    int e = blockIdx.x * blockDim.x + threadIdx.x;
    if (e < E) atomicAdd(&deg[dst[e]], 1);
}
__global__ void k_dis(const int* __restrict__ deg, float* __restrict__ dis, int n) {
    int i = blockIdx.x * blockDim.x + threadIdx.x;
    if (i < n) dis[i] = rsqrtf((float)deg[i]);   // deg >= 1 guaranteed
}

// OUT[n,c] = H[n,c] * dis[n]^2   (self-loop message; also initializes OUT)
__global__ void k_selfloop(const float* __restrict__ H, const float* __restrict__ dis,
                           float* __restrict__ OUT, int Nn, int C) {
    size_t i = (size_t)blockIdx.x * blockDim.x + threadIdx.x;
    if (i >= (size_t)Nn * C) return;
    int n   = (int)(i / (size_t)C);
    float d = dis[n];
    OUT[i]  = H[i] * d * d;
}

// One wave per edge; lanes stride channels. Hardware fp32 global atomics.
__global__ void k_edge_scatter(const float* __restrict__ H, const float* __restrict__ dis,
                               const int* __restrict__ src, const int* __restrict__ dst,
                               float* __restrict__ OUT, int E, int C) {
    int w    = (int)(((size_t)blockIdx.x * blockDim.x + threadIdx.x) >> 5);
    int lane = threadIdx.x & 31;
    if (w >= E) return;
    int s = src[w], d = dst[w];
    float nrm = dis[s] * dis[d];
    const float* hs = H + (size_t)s * C;
    float* od       = OUT + (size_t)d * C;
    for (int c = lane; c < C; c += 32)
        __hip_atomic_fetch_add(&od[c], hs[c] * nrm,
                               __ATOMIC_RELAXED, __HIP_MEMORY_SCOPE_AGENT);
}

// OUT = relu(OUT + b[c]); C is a power of two here.
__global__ void k_bias_relu(float* __restrict__ OUT, const float* __restrict__ bias,
                            size_t total, int cmask) {
    size_t i = (size_t)blockIdx.x * blockDim.x + threadIdx.x;
    if (i >= total) return;
    OUT[i] = fmaxf(OUT[i] + bias[i & (size_t)cmask], 0.0f);
}

// ---------------------------------------------------------------------------
// Global max pool (post-ReLU values >= 0 -> int-compare atomicMax valid)
// ---------------------------------------------------------------------------
__global__ void k_pool_init(float* __restrict__ pool, int n) {
    int i = blockIdx.x * blockDim.x + threadIdx.x;
    if (i < n) pool[i] = 0.0f;
}
__global__ void k_pool_max(const float* __restrict__ h, const int* __restrict__ batch,
                           float* __restrict__ pool, int Nn) {
    int w    = (int)(((size_t)blockIdx.x * blockDim.x + threadIdx.x) >> 5);
    int lane = threadIdx.x & 31;
    if (w >= Nn) return;
    int g   = batch[w];
    float v = h[(size_t)w * 32 + lane];
    atomicMax((int*)&pool[(size_t)g * 32 + lane], __float_as_int(v));
}

// out[g] = dot(last[g,:], Wout) + bout
__global__ void k_final(const float* __restrict__ last, const float* __restrict__ Wout,
                        const float* __restrict__ bout, float* __restrict__ out,
                        int G, int K) {
    int g = blockIdx.x * blockDim.x + threadIdx.x;
    if (g >= G) return;
    float s = bout[0];
    const float* row = last + (size_t)g * K;
    for (int k = 0; k < K; ++k) s += row[k] * Wout[k];
    out[g] = s;
}

// ---------------------------------------------------------------------------
extern "C" void kernel_launch(void* const* d_in, const int* in_sizes, int n_in,
                              void* d_out, int out_size, void* d_ws, size_t ws_size,
                              hipStream_t stream) {
    const float* x    = (const float*)d_in[0];
    const int*   eidx = (const int*)d_in[1];
    const int*   bat  = (const int*)d_in[2];
    const float* W1 = (const float*)d_in[3];  const float* b1 = (const float*)d_in[4];
    const float* W2 = (const float*)d_in[5];  const float* b2 = (const float*)d_in[6];
    const float* W3 = (const float*)d_in[7];  const float* b3 = (const float*)d_in[8];
    const float* W4 = (const float*)d_in[9];  const float* b4 = (const float*)d_in[10];
    const float* Wg1 = (const float*)d_in[11]; const float* bg1 = (const float*)d_in[12];
    const float* Wg2 = (const float*)d_in[13]; const float* bg2 = (const float*)d_in[14];
    const float* Wg3 = (const float*)d_in[15]; const float* bg3 = (const float*)d_in[16];
    const float* Wout = (const float*)d_in[17]; const float* bout = (const float*)d_in[18];

    const int Nn = in_sizes[2];          // 100000
    const int E  = in_sizes[1] / 2;      // 3200000
    const int F  = in_sizes[0] / Nn;     // 78
    const int G  = out_size / 257;       // 2048
    const int C1 = in_sizes[4], C2 = in_sizes[6], C3 = in_sizes[8], C4 = in_sizes[10];
    const int M1 = in_sizes[12], M2 = in_sizes[14], M3 = in_sizes[16];

    const int* src = eidx;
    const int* dst = eidx + E;

    // workspace carve-up (256B aligned)
    char* ws = (char*)d_ws;
    size_t off = 0;
    auto alloc = [&](size_t bytes) -> void* {
        void* p = ws + off;
        off += (bytes + 255) & ~(size_t)255;
        return p;
    };
    float* dis  = (float*)alloc((size_t)Nn * sizeof(float));
    int*   degi = (int*)  alloc((size_t)Nn * sizeof(int));
    float* bufA = (float*)alloc((size_t)Nn * 256 * sizeof(float));  // H (x@W)
    float* bufB = (float*)alloc((size_t)Nn * 256 * sizeof(float));  // OUT / next X
    float* pool = (float*)alloc((size_t)G * 32 * sizeof(float));
    float* g1   = (float*)alloc((size_t)G * (size_t)M1 * sizeof(float));
    float* g2   = (float*)alloc((size_t)G * (size_t)M2 * sizeof(float));
    (void)ws_size; (void)n_in;

    const int T = 256;

    // degrees + inverse-sqrt norm
    k_deg_init <<<(Nn + T - 1) / T, T, 0, stream>>>(degi, Nn);
    k_deg_accum<<<(E  + T - 1) / T, T, 0, stream>>>(dst, degi, E);
    k_dis      <<<(Nn + T - 1) / T, T, 0, stream>>>(degi, dis, Nn);

    // GCN layer: H = X@W (WMMA); OUT = self + scatter; OUT = relu(OUT + b)
    auto gcn = [&](const float* X, const float* W, const float* b, int K, int C) {
        dim3 grid((Nn + BM - 1) / BM, C / BN);
        k_gemm_wmma<<<grid, GEMM_THREADS, 0, stream>>>(X, W, nullptr, bufA, Nn, K, C, 0);
        size_t tot = (size_t)Nn * C;
        k_selfloop<<<(unsigned)((tot + T - 1) / T), T, 0, stream>>>(bufA, dis, bufB, Nn, C);
        size_t thr = (size_t)E * 32;
        k_edge_scatter<<<(unsigned)((thr + T - 1) / T), T, 0, stream>>>(bufA, dis, src, dst, bufB, E, C);
        k_bias_relu<<<(unsigned)((tot + T - 1) / T), T, 0, stream>>>(bufB, b, tot, C - 1);
    };
    gcn(x,    W1, b1, F,  C1);   // 78  -> 256
    gcn(bufB, W2, b2, C1, C2);   // 256 -> 128
    gcn(bufB, W3, b3, C2, C3);   // 128 -> 64
    gcn(bufB, W4, b4, C3, C4);   // 64  -> 32

    // global max pool -> [G, 32]
    k_pool_init<<<(G * 32 + T - 1) / T, T, 0, stream>>>(pool, G * 32);
    {
        size_t thr = (size_t)Nn * 32;
        k_pool_max<<<(unsigned)((thr + T - 1) / T), T, 0, stream>>>(bufB, bat, pool, Nn);
    }

    // MLP head (bias+relu fused in WMMA epilogue); `last` lands in d_out
    float* outp = (float*)d_out;
    float* last = outp + G;
    k_gemm_wmma<<<dim3((G + BM - 1) / BM, M1 / BN), GEMM_THREADS, 0, stream>>>(
        pool, Wg1, bg1, g1, G, C4, M1, 1);                 // 32   -> 1024
    k_gemm_wmma<<<dim3((G + BM - 1) / BM, M2 / BN), GEMM_THREADS, 0, stream>>>(
        g1, Wg2, bg2, g2, G, M1, M2, 1);                   // 1024 -> 512
    k_gemm_wmma<<<dim3((G + BM - 1) / BM, M3 / BN), GEMM_THREADS, 0, stream>>>(
        g2, Wg3, bg3, last, G, M2, M3, 1);                 // 512  -> 256

    k_final<<<(G + T - 1) / T, T, 0, stream>>>(last, Wout, bout, outp, G, M3);
}
